// Window_Attention_73701638800172
// MI455X (gfx1250) — compile-verified, matrix operand order checked by reading, not run on hardware
//
#include <hip/hip_runtime.h>
#include <hip/hip_bf16.h>
#include <stdint.h>

typedef __attribute__((ext_vector_type(16))) __bf16 v16bf;
typedef __attribute__((ext_vector_type(8)))  float  v8f;

#define LANE (threadIdx.x & 31)
#define WAVE (threadIdx.x >> 5)

// ---------------------------------------------------------------------------
// Fragment helpers — 16-bit A/B layout per CDNA5 ISA 7.12.2:
//   lane&15 -> row (A) / col (B);  lane>>4 -> K-half;  elements map to
//   k = kb+e (e<8) and k = 16+kb+(e-8), kb = 8*(lane>>4).
// Source tile in LDS is row-major with K contiguous -> two ds_load_b128/lane.
// ---------------------------------------------------------------------------
__device__ inline v16bf load_frag(const __bf16* p, int stride) {
  const int lane = threadIdx.x & 31;
  const int row  = lane & 15;
  const int kb   = (lane >> 4) << 3;
  const __bf16* r0 = p + row * stride + kb;
  v16bf f;
#pragma unroll
  for (int e = 0; e < 8; ++e) { f[e] = r0[e]; f[e + 8] = r0[16 + e]; }
  return f;
}

__device__ inline v8f wmma_bf16(v16bf a, v16bf b, v8f c) {
  return __builtin_amdgcn_wmma_f32_16x16x32_bf16(false, a, false, b,
                                                 (short)0, c, false, false);
}

// CDNA5 async global->LDS copy (ASYNCcnt-tracked), GV mode (64-bit vaddr).
// Generic LDS pointer's low 32 bits are the raw LDS offset (ISA 10.2).
__device__ inline void async_ld_b128(unsigned lds_off, const void* gaddr) {
  asm volatile("global_load_async_to_lds_b128 %0, %1, off"
               :: "v"(lds_off), "v"((uint64_t)(uintptr_t)gaddr)
               : "memory");
}
__device__ inline void wait_async_le4() {
  asm volatile("s_wait_asynccnt 4" ::: "memory");
}
__device__ inline void wait_async_le0() {
  asm volatile("s_wait_asynccnt 0" ::: "memory");
}

// window-ordered row r -> (batch, pixel row, pixel col)
__device__ inline void r2pix(int r, int& b, int& pr, int& pc) {
  int n = r & 63, wdw = r >> 6;
  b = wdw >> 6;
  int wr = wdw & 63, wh = wr >> 3, ww = wr & 7, i = n >> 3, j = n & 7;
  pr = wh * 8 + i;
  pc = ww * 8 + j;
}

// ---------------------------------------------------------------------------
// One-shot fp32 -> bf16 conversion of all four weight matrices (1.5 MiB,
// stays L2-resident) so every GEMM operand can use the async-to-LDS path.
// ---------------------------------------------------------------------------
__global__ void cvt_w_kernel(const float* __restrict__ a, int na,
                             const float* __restrict__ b, int nb,
                             const float* __restrict__ c, int nc,
                             const float* __restrict__ d, int nd,
                             __bf16* __restrict__ out) {
  int i = blockIdx.x * 256 + threadIdx.x;
  float v;
  if (i < na) v = a[i];
  else if (i < na + nb) v = b[i - na];
  else if (i < na + nb + nc) v = c[i - na - nb];
  else if (i < na + nb + nc + nd) v = d[i - na - nb - nc];
  else return;
  out[i] = (__bf16)v;
}

// ---------------------------------------------------------------------------
// LN1: NCHW fp32 -> window-ordered [r][256] bf16. One thread per token.
// ---------------------------------------------------------------------------
__global__ void ln1_kernel(const float* __restrict__ x,
                           const float* __restrict__ g,
                           const float* __restrict__ bt,
                           __bf16* __restrict__ xn) {
  int r = blockIdx.x * 256 + threadIdx.x;
  int b, pr, pc; r2pix(r, b, pr, pc);
  const float* px = x + (size_t)b * 256 * 4096 + pr * 64 + pc;
  float s = 0.f, s2 = 0.f;
#pragma unroll 8
  for (int ch = 0; ch < 256; ++ch) {
    float v = px[(size_t)ch * 4096];
    s += v; s2 += v * v;
  }
  float mean = s * (1.f / 256.f);
  float var  = s2 * (1.f / 256.f) - mean * mean;
  float rstd = rsqrtf(var + 1e-5f);
  __bf16* out = xn + (size_t)r * 256;
#pragma unroll 8
  for (int ch = 0; ch < 256; ++ch) {
    float v = px[(size_t)ch * 4096];
    out[ch] = (__bf16)((v - mean) * rstd * g[ch] + bt[ch]);
  }
}

// ---------------------------------------------------------------------------
// LN2: [r][256] fp32 contiguous -> bf16. Wave per token, shfl reduce (w32).
// ---------------------------------------------------------------------------
__global__ void ln2_kernel(const float* __restrict__ xf,
                           const float* __restrict__ g,
                           const float* __restrict__ bt,
                           __bf16* __restrict__ h) {
  int r = blockIdx.x * 8 + WAVE;
  const float* px = xf + (size_t)r * 256;
  float v[8]; float s = 0.f;
#pragma unroll
  for (int k = 0; k < 8; ++k) { v[k] = px[LANE * 8 + k]; s += v[k]; }
#pragma unroll
  for (int m = 1; m < 32; m <<= 1) s += __shfl_xor(s, m, 32);
  float mean = s * (1.f / 256.f);
  float vs = 0.f;
#pragma unroll
  for (int k = 0; k < 8; ++k) { float d = v[k] - mean; vs += d * d; }
#pragma unroll
  for (int m = 1; m < 32; m <<= 1) vs += __shfl_xor(vs, m, 32);
  float rstd = rsqrtf(vs * (1.f / 256.f) + 1e-5f);
  __bf16* out = h + (size_t)r * 256;
#pragma unroll
  for (int k = 0; k < 8; ++k) {
    int ch = LANE * 8 + k;
    out[ch] = (__bf16)((v[k] - mean) * rstd * g[ch] + bt[ch]);
  }
}

// ---------------------------------------------------------------------------
// 64x64-tile WMMA GEMM, async double-buffered: C = A[row][k] * Bw[col][k]^T.
// A, Bw: bf16 row-major stride KDIM. Tiles stream into LDS via
// GLOBAL_LOAD_ASYNC_TO_LDS_B128 (ASYNCcnt), next tile in flight during WMMA.
// 128 threads = 4 waves; wave w owns rows w*16..w*16+15 (4 N-subtiles).
//   MODE 0: qkv  (+bias, q-part *4.0, bf16 out ld=768)
//   MODE 1: proj (+bias + NCHW shortcut extra0, fp32 out ld=256)
//   MODE 2: fc1  (+bias, exact GELU, bf16 out ld=1024)
//   MODE 3: fc2  (+bias + residual extra0[r*256+c], fp32 NCHW store)
// ---------------------------------------------------------------------------
template <int KDIM, int MODE>
__global__ void gemm_kernel(const __bf16* __restrict__ A,
                            const __bf16* __restrict__ Bw,
                            const float* __restrict__ bias,
                            __bf16* __restrict__ outb,
                            float* __restrict__ outf,
                            const float* __restrict__ extra0) {
  __shared__ __bf16 As[2][64][32];
  __shared__ __bf16 Bs[2][64][32];
  const int tid = threadIdx.x;
  const int row0 = blockIdx.x * 64, col0 = blockIdx.y * 64;
  const unsigned ldsA = (unsigned)(uintptr_t)&As[0][0][0];
  const unsigned ldsB = (unsigned)(uintptr_t)&Bs[0][0][0];
  constexpr int NK = KDIM / 32;

  // issue one 64x32 bf16 tile (A and B) into LDS buffer `buf`:
  // 2 x 16-byte chunks per thread per matrix = 4 async ops per wave.
  auto issue_tile = [&](int kt, int buf) {
#pragma unroll
    for (int c = 0; c < 2; ++c) {
      int chunk = tid + 128 * c;              // 0..255
      int row = chunk >> 2, col = (chunk & 3) * 8;
      unsigned loff = (unsigned)((row * 32 + col) * 2 + buf * 4096);
      async_ld_b128(ldsA + loff,
                    A + (size_t)(row0 + row) * KDIM + kt * 32 + col);
      async_ld_b128(ldsB + loff,
                    Bw + (size_t)(col0 + row) * KDIM + kt * 32 + col);
    }
  };

  v8f z = {0.f, 0.f, 0.f, 0.f, 0.f, 0.f, 0.f, 0.f};
  v8f acc[4] = {z, z, z, z};

  issue_tile(0, 0);
  for (int kt = 0; kt < NK; ++kt) {
    const int buf = kt & 1;
    if (kt + 1 < NK) {
      issue_tile(kt + 1, buf ^ 1);
      wait_async_le4();   // in-order completion => tile kt resident
    } else {
      wait_async_le0();
    }
    __syncthreads();
    v16bf af = load_frag(&As[buf][WAVE * 16][0], 32);
#pragma unroll
    for (int j = 0; j < 4; ++j) {
      v16bf bf = load_frag(&Bs[buf][j * 16][0], 32);
      acc[j] = wmma_bf16(af, bf, acc[j]);
    }
    __syncthreads();    // protect buffer before it is overwritten
  }

  const int hi = LANE >> 4, lo = LANE & 15;
#pragma unroll
  for (int j = 0; j < 4; ++j) {
    int c = col0 + j * 16 + lo;
    float bv = bias[c];
#pragma unroll
    for (int r = 0; r < 8; ++r) {
      int row = row0 + WAVE * 16 + r + 8 * hi;
      float val = acc[j][r] + bv;
      if constexpr (MODE == 0) {
        if (c < 256) val *= 4.0f;  // qk_scale on Q
        outb[(size_t)row * 768 + c] = (__bf16)val;
      } else if constexpr (MODE == 1) {
        int b, pr, pc; r2pix(row, b, pr, pc);
        val += extra0[((size_t)b * 256 + c) * 4096 + pr * 64 + pc];
        outf[(size_t)row * 256 + c] = val;
      } else if constexpr (MODE == 2) {
        val = 0.5f * val * (1.0f + erff(val * 0.70710678118654752f));
        outb[(size_t)row * 1024 + c] = (__bf16)val;
      } else {
        val += extra0[(size_t)row * 256 + c];
        int b, pr, pc; r2pix(row, b, pr, pc);
        outf[(((size_t)b * 256 + c) * 64 + pr) * 64 + pc] = val;
      }
    }
  }
}

// ---------------------------------------------------------------------------
// Attention: one 128-thread block per (window, head). S = qk^T (4 WMMA/wave),
// +rel-pos bias, shfl softmax (lane&15 spread + 4 N-subtiles), then
// P (bf16, per-wave LDS) @ V^T (4 WMMA/wave) -> fp32 ao.
// ---------------------------------------------------------------------------
__global__ void attn_kernel(const __bf16* __restrict__ qkv,
                            const float* __restrict__ rpb,
                            float* __restrict__ ao) {
  __shared__ __bf16 qs[64][32];
  __shared__ __bf16 ks_[64][32];
  __shared__ __bf16 vt[32][64];      // v transposed: [d][n]
  __shared__ __bf16 ps[4][16][64];   // per-wave P rows
  __shared__ float  bias_s[64][64];
  const int tid = threadIdx.x;
  const int wdw = blockIdx.x >> 3, head = blockIdx.x & 7;
  const size_t base = (size_t)wdw * 64 * 768 + head * 32;

  for (int idx = tid; idx < 2048; idx += 128) {
    int n = idx >> 5, d = idx & 31;
    qs[n][d]  = qkv[base + (size_t)n * 768 + d];
    ks_[n][d] = qkv[base + (size_t)n * 768 + 256 + d];
    vt[d][n]  = qkv[base + (size_t)n * 768 + 512 + d];
  }
  for (int idx = tid; idx < 4096; idx += 128) {
    int n = idx >> 6, m = idx & 63;
    int t = ((n >> 3) - (m >> 3) + 7) * 15 + ((n & 7) - (m & 7) + 7);
    bias_s[n][m] = rpb[t * 8 + head];
  }
  __syncthreads();

  const int hi = LANE >> 4, lo = LANE & 15;
  v8f z = {0.f, 0.f, 0.f, 0.f, 0.f, 0.f, 0.f, 0.f};

  // S = q @ k^T for this wave's 16 query rows
  v8f accs[4] = {z, z, z, z};
  v16bf aq = load_frag(&qs[WAVE * 16][0], 32);
#pragma unroll
  for (int j = 0; j < 4; ++j) {
    v16bf bk = load_frag(&ks_[j * 16][0], 32);
    accs[j] = wmma_bf16(aq, bk, accs[j]);
  }

  // softmax over 64 keys
  float e[4][8], mx[8], sm[8];
#pragma unroll
  for (int r = 0; r < 8; ++r) mx[r] = -3.0e38f;
#pragma unroll
  for (int j = 0; j < 4; ++j)
#pragma unroll
    for (int r = 0; r < 8; ++r) {
      int qi = WAVE * 16 + r + 8 * hi;
      float v = accs[j][r] + bias_s[qi][j * 16 + lo];
      e[j][r] = v;
      mx[r] = fmaxf(mx[r], v);
    }
#pragma unroll
  for (int r = 0; r < 8; ++r)
#pragma unroll
    for (int m = 1; m < 16; m <<= 1) mx[r] = fmaxf(mx[r], __shfl_xor(mx[r], m, 32));
#pragma unroll
  for (int r = 0; r < 8; ++r) sm[r] = 0.f;
#pragma unroll
  for (int j = 0; j < 4; ++j)
#pragma unroll
    for (int r = 0; r < 8; ++r) {
      float ex = __expf(e[j][r] - mx[r]);
      e[j][r] = ex;
      sm[r] += ex;
    }
#pragma unroll
  for (int r = 0; r < 8; ++r) {
#pragma unroll
    for (int m = 1; m < 16; m <<= 1) sm[r] += __shfl_xor(sm[r], m, 32);
    sm[r] = 1.0f / sm[r];
  }
#pragma unroll
  for (int j = 0; j < 4; ++j)
#pragma unroll
    for (int r = 0; r < 8; ++r)
      ps[WAVE][r + 8 * hi][j * 16 + lo] = (__bf16)(e[j][r] * sm[r]);

  // O = P @ V  (K = 64 tokens in two 32-steps; N = 32 dims in two subtiles)
  v8f acco[2] = {z, z};
#pragma unroll
  for (int ks2 = 0; ks2 < 2; ++ks2) {
    v16bf ap = load_frag(&ps[WAVE][0][ks2 * 32], 64);
#pragma unroll
    for (int j2 = 0; j2 < 2; ++j2) {
      v16bf bv = load_frag(&vt[j2 * 16][ks2 * 32], 64);
      acco[j2] = wmma_bf16(ap, bv, acco[j2]);
    }
  }
#pragma unroll
  for (int j2 = 0; j2 < 2; ++j2)
#pragma unroll
    for (int r = 0; r < 8; ++r) {
      int qi = WAVE * 16 + r + 8 * hi;
      ao[((size_t)wdw * 64 + qi) * 256 + head * 32 + j2 * 16 + lo] = acco[j2][r];
    }
}

// ---------------------------------------------------------------------------
// Gaze: 9x9 depthwise conv on the dilated fold of V (bf16 v from qkv),
// fused with (attn_out + gz + bias) -> bf16 for the proj GEMM.
// ---------------------------------------------------------------------------
__global__ void gaze_kernel(const __bf16* __restrict__ qkv,
                            const float* __restrict__ ao,
                            const float* __restrict__ gw,
                            const float* __restrict__ gb,
                            __bf16* __restrict__ aobf) {
  const int ch = threadIdx.x;
  const int pix = blockIdx.x;
  const int b = pix >> 12, R = (pix >> 6) & 63, Cc = pix & 63;
  float acc = 0.f;
#pragma unroll
  for (int dy = 0; dy < 9; ++dy) {
    int Rp = R + dy - 4;
    if ((unsigned)Rp >= 64u) continue;
#pragma unroll
    for (int dx = 0; dx < 9; ++dx) {
      int Cp = Cc + dx - 4;
      if ((unsigned)Cp >= 64u) continue;
      int i = Rp >> 3, wh = Rp & 7, j = Cp >> 3, ww = Cp & 7;
      size_t rs = ((size_t)(b * 64 + wh * 8 + ww)) * 64 + (i * 8 + j);
      acc += gw[(dy * 9 + dx) * 256 + ch] * (float)qkv[rs * 768 + 512 + ch];
    }
  }
  int i = R >> 3, wh = R & 7, j = Cc >> 3, ww = Cc & 7;
  size_t ro = ((size_t)(b * 64 + wh * 8 + ww)) * 64 + (i * 8 + j);
  aobf[ro * 256 + ch] = (__bf16)(ao[ro * 256 + ch] + acc + gb[ch]);
}

// ---------------------------------------------------------------------------
extern "C" void kernel_launch(void* const* d_in, const int* in_sizes, int n_in,
                              void* d_out, int out_size, void* d_ws, size_t ws_size,
                              hipStream_t stream) {
  const float* x       = (const float*)d_in[2];
  const float* norm1_g = (const float*)d_in[3];
  const float* norm1_b = (const float*)d_in[4];
  const float* qkv_w   = (const float*)d_in[5];
  const float* qkv_b   = (const float*)d_in[6];
  const float* rpb     = (const float*)d_in[7];
  const float* proj_w  = (const float*)d_in[8];
  const float* proj_b  = (const float*)d_in[9];
  const float* gaze_w  = (const float*)d_in[10];
  const float* gaze_b  = (const float*)d_in[11];
  const float* norm2_g = (const float*)d_in[12];
  const float* norm2_b = (const float*)d_in[13];
  const float* fc1_w   = (const float*)d_in[14];
  const float* fc1_b   = (const float*)d_in[15];
  const float* fc2_w   = (const float*)d_in[16];
  const float* fc2_b   = (const float*)d_in[17];

  char* ws = (char*)d_ws;
  __bf16* xn   = (__bf16*)ws;                      // [65536][256] bf16   32MiB
  __bf16* qkvb = (__bf16*)(ws + (32ull  << 20));   // [65536][768] bf16   96MiB
  float*  ao   = (float*) (ws + (128ull << 20));   // [65536][256] f32    64MiB
  __bf16* aobf = (__bf16*)(ws + (192ull << 20));   // [65536][256] bf16   32MiB
  float*  xf   = (float*) (ws + (224ull << 20));   // [65536][256] f32    64MiB
  __bf16* hbf  = (__bf16*)ws;                      // reuse xn region
  __bf16* fc1o = (__bf16*)(ws + (32ull  << 20));   // reuse qkv region 128MiB
  __bf16* wb   = (__bf16*)(ws + (288ull << 20));   // bf16 weights     1.5MiB
  __bf16* qkv_wb  = wb;                 // 768*256  = 196608
  __bf16* proj_wb = wb + 196608;        // 256*256  =  65536
  __bf16* fc1_wb  = wb + 262144;        // 1024*256 = 262144
  __bf16* fc2_wb  = wb + 524288;        // 256*1024 = 262144

  cvt_w_kernel<<<3072, 256, 0, stream>>>(qkv_w, 196608, proj_w, 65536,
                                         fc1_w, 262144, fc2_w, 262144, wb);

  ln1_kernel<<<256, 256, 0, stream>>>(x, norm1_g, norm1_b, xn);

  gemm_kernel<256, 0><<<dim3(1024, 12), 128, 0, stream>>>(
      xn, qkv_wb, qkv_b, qkvb, nullptr, nullptr);

  attn_kernel<<<8192, 128, 0, stream>>>(qkvb, rpb, ao);

  gaze_kernel<<<65536, 256, 0, stream>>>(qkvb, ao, gaze_w, gaze_b, aobf);

  gemm_kernel<256, 1><<<dim3(1024, 4), 128, 0, stream>>>(
      aobf, proj_wb, proj_b, nullptr, xf, x);

  ln2_kernel<<<8192, 256, 0, stream>>>(xf, norm2_g, norm2_b, hbf);

  gemm_kernel<256, 2><<<dim3(1024, 16), 128, 0, stream>>>(
      hbf, fc1_wb, fc1_b, fc1o, nullptr, nullptr);

  gemm_kernel<1024, 3><<<dim3(1024, 4), 128, 0, stream>>>(
      fc1o, fc2_wb, fc2_b, nullptr, (float*)d_out, xf);
}
